// ProjBlock3_58025008169495
// MI455X (gfx1250) — matrix operand order repaired and checked by reading, not compiled, not run on hardware
//
#include <hip/hip_runtime.h>

// ---------------- CDNA5 WMMA types ----------------
typedef __attribute__((ext_vector_type(16))) __bf16 v16bf;
typedef __attribute__((ext_vector_type(8)))  __bf16 v8bf;
typedef __attribute__((ext_vector_type(8)))  float  v8f;

#define B_  8
#define C_  64
#define H_  256
#define W_  256
#define HW_ 65536

static __device__ inline v8f wmma_bf16(v16bf a, v16bf b, v8f c) {
  return __builtin_amdgcn_wmma_f32_16x16x32_bf16(false, a, false, b, (short)0, c, false, false);
}

// A-tile (16x32 bf16): lane<16 holds K = 0..15, lane>=16 holds K = 16..31 (sequential), M = lane&15
static __device__ inline v16bf load_a_f32(const float* src, int ld, int k0, float s) {
  int lane = threadIdx.x & 31;
  int m = lane & 15;
  int kb = (lane & 16) ? 16 : 0;
  const float* p = src + m * ld + k0 + kb;
  v16bf a;
#pragma unroll
  for (int j = 0; j < 8; ++j) {
    a[j]     = (__bf16)(p[j] * s);
    a[j + 8] = (__bf16)(p[j + 8] * s);
  }
  return a;
}

static __device__ inline v16bf load_a_bf16(const __bf16* src, int ld, int k0) {
  int lane = threadIdx.x & 31;
  int m = lane & 15;
  int kb = (lane & 16) ? 16 : 0;
  const __bf16* p = src + m * ld + k0 + kb;      // 16B aligned
  v8bf lo = *(const v8bf*)p;
  v8bf hi = *(const v8bf*)(p + 8);
  v16bf a;
#pragma unroll
  for (int j = 0; j < 8; ++j) { a[j] = lo[j]; a[j + 8] = hi[j]; }
  return a;
}

// B-tile (32x16 bf16): N = lane&15, lane<16 holds K=0..15, lane>=16 holds K=16..31 (sequential)
// source is row-major over (N, K): element (k, n) at base[(n0+n)*ld + k0 + k]
static __device__ inline v16bf load_b_bf16(const __bf16* base, int ld, int n0, int k0) {
  int lane = threadIdx.x & 31;
  int n = lane & 15;
  int kb = (lane & 16) ? 16 : 0;
  const __bf16* p = base + (size_t)(n0 + n) * ld + k0 + kb;  // 32B aligned
  v8bf lo = *(const v8bf*)p;
  v8bf hi = *(const v8bf*)(p + 8);
  v16bf b;
#pragma unroll
  for (int j = 0; j < 8; ++j) { b[j] = lo[j]; b[j + 8] = hi[j]; }
  return b;
}

// B-tile from x (f32, channel-major): element (k=i, n=p) at xb[i*HW + p0 + n]; coalesced across lanes
static __device__ inline v16bf load_b_x(const float* xb, int k0, int p0) {
  int lane = threadIdx.x & 31;
  int n = lane & 15;
  int kb = (lane & 16) ? 16 : 0;
  const float* p = xb + (size_t)(k0 + kb) * HW_ + p0 + n;
  v16bf b;
#pragma unroll
  for (int j = 0; j < 16; ++j) b[j] = (__bf16)p[(size_t)j * HW_];
  return b;
}

// ---------------- Kernel 1: fused QKV 1x1-conv, bf16 outputs, V stored transposed ----------------
// grid = B * 512 blocks of 256 threads; each wave handles 16 consecutive pixels, all 64 out-channels
__global__ __launch_bounds__(256, 1) void qkv_proj_kernel(
    const float* __restrict__ x,
    const float* __restrict__ Wq, const float* __restrict__ bq,
    const float* __restrict__ Wk, const float* __restrict__ bk,
    const float* __restrict__ Wv, const float* __restrict__ bv,
    __bf16* __restrict__ Qo, __bf16* __restrict__ Ko, __bf16* __restrict__ Vto) {
  const int wid  = threadIdx.x >> 5;
  const int lane = threadIdx.x & 31;
  const int b    = blockIdx.x >> 9;            // 512 blocks per batch
  const int p0   = (blockIdx.x & 511) * 128 + wid * 16;
  const float* xb = x + (size_t)b * C_ * HW_;
  const int n    = lane & 15;
  const int roff = (lane & 16) ? 8 : 0;

  const float* Ws[3] = {Wq, Wk, Wv};
  const float* bs[3] = {bq, bk, bv};

  v8f acc[3][4];
#pragma unroll
  for (int pr = 0; pr < 3; ++pr) {
    const float sc = (pr == 0) ? 0.25f : 1.0f;   // fold 1/H^0.25 into Q path
#pragma unroll
    for (int ot = 0; ot < 4; ++ot)
#pragma unroll
      for (int v = 0; v < 8; ++v)
        acc[pr][ot][v] = bs[pr][ot * 16 + v + roff] * sc;
  }

#pragma unroll
  for (int kt = 0; kt < 2; ++kt) {             // K-dim = 64 input channels, 2 steps of 32
    v16bf bx = load_b_x(xb, kt * 32, p0);
#pragma unroll
    for (int pr = 0; pr < 3; ++pr) {
      const float sc = (pr == 0) ? 0.25f : 1.0f;
#pragma unroll
      for (int ot = 0; ot < 4; ++ot) {
        v16bf a = load_a_f32(Ws[pr] + ot * 16 * C_, C_, kt * 32, sc);
        acc[pr][ot] = wmma_bf16(a, bx, acc[pr][ot]);
      }
    }
  }

  const size_t base = (size_t)b * C_ * HW_;
  const int h  = p0 >> 8;                      // 16-pixel tile never crosses a row (W=256)
  const int w0 = p0 & 255;
#pragma unroll
  for (int ot = 0; ot < 4; ++ot) {
#pragma unroll
    for (int v = 0; v < 8; ++v) {
      const int o = ot * 16 + v + roff;
      Qo[base + (size_t)o * HW_ + p0 + n]          = (__bf16)acc[0][ot][v];
      Ko[base + (size_t)o * HW_ + p0 + n]          = (__bf16)acc[1][ot][v];
      Vto[base + (size_t)o * HW_ + (w0 + n) * H_ + h] = (__bf16)acc[2][ot][v];  // transposed
    }
  }
}

// ---------------- Kernel 2: per-(b,c) row attention + softmax + BN + PReLU ----------------
// grid = B*C*2 blocks of 256 threads; each wave owns a 16-row Q strip
__global__ __launch_bounds__(256, 1) void attn_kernel(
    const __bf16* __restrict__ Qb, const __bf16* __restrict__ Kb, const __bf16* __restrict__ Vtb,
    const float* __restrict__ gamma, const float* __restrict__ beta,
    const float* __restrict__ rmean, const float* __restrict__ rvar,
    const float* __restrict__ alpha, float* __restrict__ out) {
  __shared__ __bf16 Pl[8][16][256];            // 64 KB of the 320 KB WGP LDS

  const int wid  = threadIdx.x >> 5;
  const int lane = threadIdx.x & 31;
  const int bc   = blockIdx.x >> 1;            // b*64 + c
  const int c    = bc & 63;
  const int m0   = (blockIdx.x & 1) * 128 + wid * 16;
  const int n    = lane & 15;
  const int roff = (lane & 16) ? 8 : 0;

  const __bf16* Qp = Qb  + (size_t)bc * HW_;
  const __bf16* Kp = Kb  + (size_t)bc * HW_;
  const __bf16* Vp = Vtb + (size_t)bc * HW_;

  // Q strip A-tiles, reused across all 16 score column tiles
  v16bf aq[8];
#pragma unroll
  for (int kt = 0; kt < 8; ++kt) aq[kt] = load_a_bf16(Qp + (size_t)m0 * W_, W_, kt * 32);

  // S = (Q/scale) * K^T : 16 x 256 f32 strip in registers
  v8f acc[16];
#pragma unroll
  for (int nt = 0; nt < 16; ++nt) {
    v8f a = {0.f, 0.f, 0.f, 0.f, 0.f, 0.f, 0.f, 0.f};
#pragma unroll
    for (int kt = 0; kt < 8; ++kt) {
      v16bf bt = load_b_bf16(Kp, W_, nt * 16, kt * 32);
      a = wmma_bf16(aq[kt], bt, a);
    }
    acc[nt] = a;
  }

  // softmax over the 256-wide rows (row v lives in one lane-half across 16 tiles)
#pragma unroll
  for (int v = 0; v < 8; ++v) {
    float mx = -3.4e38f;
#pragma unroll
    for (int nt = 0; nt < 16; ++nt) mx = fmaxf(mx, acc[nt][v]);
#pragma unroll
    for (int msk = 1; msk < 16; msk <<= 1) mx = fmaxf(mx, __shfl_xor(mx, msk, 32));
    float sum = 0.f;
#pragma unroll
    for (int nt = 0; nt < 16; ++nt) {
      float e = __expf(acc[nt][v] - mx);
      acc[nt][v] = e;
      sum += e;
    }
#pragma unroll
    for (int msk = 1; msk < 16; msk <<= 1) sum += __shfl_xor(sum, msk, 32);
    const float r = 1.0f / sum;
#pragma unroll
    for (int nt = 0; nt < 16; ++nt) acc[nt][v] *= r;
  }

  // C-layout -> row-major bf16 P strip in LDS (each wave touches only its own slab)
#pragma unroll
  for (int nt = 0; nt < 16; ++nt)
#pragma unroll
    for (int v = 0; v < 8; ++v)
      Pl[wid][v + roff][nt * 16 + n] = (__bf16)acc[nt][v];
  __syncthreads();

  // reload P as A-tiles
  v16bf ap[8];
#pragma unroll
  for (int kt = 0; kt < 8; ++kt) ap[kt] = load_a_bf16(&Pl[wid][0][0], 256, kt * 32);

  // BN (eval) + PReLU epilogue constants
  const float inv   = gamma[c] * rsqrtf(rvar[c] + 1e-5f);
  const float shift = beta[c] - rmean[c] * inv;
  const float al    = alpha[0];
  float* op = out + (size_t)bc * HW_ + (size_t)m0 * W_;

  // O = P * V  (B-tiles contiguous thanks to transposed V)
#pragma unroll
  for (int wt = 0; wt < 16; ++wt) {
    v8f o = {0.f, 0.f, 0.f, 0.f, 0.f, 0.f, 0.f, 0.f};
#pragma unroll
    for (int kt = 0; kt < 8; ++kt) {
      v16bf bt = load_b_bf16(Vp, H_, wt * 16, kt * 32);
      o = wmma_bf16(ap[kt], bt, o);
    }
#pragma unroll
    for (int v = 0; v < 8; ++v) {
      float t = o[v] * inv + shift;
      t = (t >= 0.f) ? t : al * t;
      op[(v + roff) * W_ + wt * 16 + n] = t;
    }
  }
}

// ---------------- launch ----------------
extern "C" void kernel_launch(void* const* d_in, const int* in_sizes, int n_in,
                              void* d_out, int out_size, void* d_ws, size_t ws_size,
                              hipStream_t stream) {
  const float* x     = (const float*)d_in[0];
  const float* Wq    = (const float*)d_in[1];
  const float* bq    = (const float*)d_in[2];
  const float* Wk    = (const float*)d_in[3];
  const float* bk    = (const float*)d_in[4];
  const float* Wv    = (const float*)d_in[5];
  const float* bv    = (const float*)d_in[6];
  const float* gamma = (const float*)d_in[7];
  const float* beta  = (const float*)d_in[8];
  const float* rmean = (const float*)d_in[9];
  const float* rvar  = (const float*)d_in[10];
  const float* alpha = (const float*)d_in[11];

  const size_t NE = (size_t)B_ * C_ * HW_;     // 33,554,432 elems
  __bf16* Qw  = (__bf16*)d_ws;
  __bf16* Kw  = Qw + NE;
  __bf16* Vtw = Kw + NE;                       // total 192 MB bf16 workspace

  qkv_proj_kernel<<<B_ * 512, 256, 0, stream>>>(x, Wq, bq, Wk, bk, Wv, bv, Qw, Kw, Vtw);
  attn_kernel<<<B_ * C_ * 2, 256, 0, stream>>>(Qw, Kw, Vtw, gamma, beta, rmean, rvar, alpha,
                                               (float*)d_out);
}
